// HierarchicalMamba_57844619543176
// MI455X (gfx1250) — matrix operand order repaired and checked by reading, not compiled, not run on hardware
//
#include <hip/hip_runtime.h>

// ---------------------------------------------------------------------------
// MI455X (gfx1250) hierarchical Mamba forward pass.
// Dense projections (in_proj, x_proj, out_proj, fus 1x1 conv) run on the bf16
// WMMA pipe (v_wmma_f32_16x16x32_bf16), one wave per 16x16 tile. All operand
// buffers are zero-padded to K multiples of 32 (and N multiples of 16) at
// pack time, so the WMMA inner loop is guard-free: 4 aligned b128 loads + 1
// wmma per K-step, no EXEC divergence. SSM scan: 16 lanes = 16 states per
// channel, half-wave __shfl_xor reduction (wave32).
// ---------------------------------------------------------------------------

typedef __bf16 bf16_t;
typedef __attribute__((ext_vector_type(16))) __bf16        v16bf;
typedef __attribute__((ext_vector_type(8)))  float         v8f;
typedef __attribute__((ext_vector_type(4)))  unsigned int  uint4v;

union Frag { v16bf v; uint4v q[2]; };

// C tiles of A[M,Kp] @ W[Npad,Kp]^T  (bf16 row-major, Kp multiple of 32,
// rows/cols beyond logical N/K are zero-padded at pack time).
// EPI 0: C[m*ldc+n]  = acc
// EPI 1: C[m*ldc+n] += acc               (residual accumulate)
// EPI 2: C[n*ldT+m]  = acc + bias[n]     (bias + transposed store, 1x1 conv)
template<int EPI>
__global__ __launch_bounds__(128)
void gemm_wmma_bf16(const bf16_t* __restrict__ A, const bf16_t* __restrict__ W,
                    float* __restrict__ C, const float* __restrict__ bias,
                    int Kp, int ldc, int ldT) {
  const int lane = threadIdx.x & 31;
  const int wave = threadIdx.x >> 5;
  const int hi   = lane >> 4;
  const int l15  = lane & 15;
  const int tm   = blockIdx.x * 4 + wave;
  const int tn   = blockIdx.y;
  const bf16_t* Arow = A + (size_t)(tm * 16 + l15) * Kp;   // A row this lane feeds
  const bf16_t* Wrow = W + (size_t)(tn * 16 + l15) * Kp;   // B column this lane feeds
  v8f acc = {0.f, 0.f, 0.f, 0.f, 0.f, 0.f, 0.f, 0.f};
  for (int k0 = 0; k0 < Kp; k0 += 32) {
    Frag a, b;
    // A 16x32 fragment: lane<16 -> k0..+7 & k0+16..+23 ; lane>=16 -> +8
    a.q[0] = *(const uint4v*)(Arow + k0 + hi * 8);
    a.q[1] = *(const uint4v*)(Arow + k0 + hi * 8 + 16);
    // B 32x16 fragment: lane<16 -> k0..+7 & k0+16..+23 ; lane>=16 -> +8
    b.q[0] = *(const uint4v*)(Wrow + k0 + hi * 8);
    b.q[1] = *(const uint4v*)(Wrow + k0 + hi * 8 + 16);
    acc = __builtin_amdgcn_wmma_f32_16x16x32_bf16(false, a.v, false, b.v,
                                                  (short)0, acc, false, false);
  }
#pragma unroll
  for (int j = 0; j < 8; ++j) {
    const int m = tm * 16 + j + hi * 8;
    const int n = tn * 16 + l15;
    if (EPI == 0)      C[(size_t)m * ldc + n] = acc[j];
    else if (EPI == 1) C[(size_t)m * ldc + n] += acc[j];
    else               C[(size_t)n * ldT + m] = acc[j] + bias[n];
  }
}

// Pack f32 (N,K) row-major -> bf16 (Npad,Kp) row-major, zero-padded.
__global__ void pack_bf16(const float* __restrict__ in, bf16_t* __restrict__ out,
                          int N, int K, int Npad, int Kp) {
  int i = blockIdx.x * blockDim.x + threadIdx.x;
  if (i >= Npad * Kp) return;
  int n = i / Kp, k = i - n * Kp;
  out[i] = (n < N && k < K) ? (bf16_t)in[(size_t)n * K + k] : (bf16_t)0.f;
}

// seq0[l,c] = in_b[c] + sum_ci in_w[c,ci] * x[ci,l]   (1x1x1 stem conv)
__global__ void init_seq(const float* __restrict__ x, const float* __restrict__ w,
                         const float* __restrict__ b, float* __restrict__ seq, int L) {
  int i = blockIdx.x * blockDim.x + threadIdx.x;
  if (i >= L * 48) return;
  int l = i / 48, c = i - l * 48;
  float acc = b[c];
#pragma unroll
  for (int ci = 0; ci < 4; ++ci) acc += w[c * 4 + ci] * x[(size_t)ci * L + l];
  seq[i] = acc;
}

// LayerNorm over d; write bf16 row of stride Kp with zeroed tail.
__global__ __launch_bounds__(256)
void layernorm_bf16(const float* __restrict__ x, const float* __restrict__ w,
                    const float* __restrict__ b, bf16_t* __restrict__ out,
                    int L, int d, int Kp) {
  const int wave = threadIdx.x >> 5, lane = threadIdx.x & 31;
  const int row  = blockIdx.x * 8 + wave;
  if (row >= L) return;
  const float* xr = x + (size_t)row * d;
  float s = 0.f, ss = 0.f;
  for (int c = lane; c < d; c += 32) { float v = xr[c]; s += v; ss += v * v; }
#pragma unroll
  for (int m = 16; m >= 1; m >>= 1) { s += __shfl_xor(s, m, 32); ss += __shfl_xor(ss, m, 32); }
  const float mu  = s / d;
  const float inv = rsqrtf(ss / d - mu * mu + 1e-5f);
  bf16_t* orow = out + (size_t)row * Kp;
  for (int c = lane; c < Kp; c += 32)
    orow[c] = (c < d) ? (bf16_t)((xr[c] - mu) * inv * w[c] + b[c]) : (bf16_t)0.f;
}

// causal depthwise conv (width 3) + SiLU on x-half of xz; write f32 + bf16
__global__ void dwconv_silu(const float* __restrict__ xz, const float* __restrict__ cw,
                            const float* __restrict__ cb, float* __restrict__ xf,
                            bf16_t* __restrict__ xb, int L, int di) {
  int i = blockIdx.x * blockDim.x + threadIdx.x;
  if (i >= L * di) return;
  int l = i / di, c = i - l * di;
  const int stride = 2 * di;
  float v = cb[c];
  v += (l >= 2 ? xz[(size_t)(l - 2) * stride + c] : 0.f) * cw[c * 3 + 0];
  v += (l >= 1 ? xz[(size_t)(l - 1) * stride + c] : 0.f) * cw[c * 3 + 1];
  v += xz[(size_t)l * stride + c] * cw[c * 3 + 2];
  float s = v / (1.f + __expf(-v));   // SiLU
  xf[i] = s;
  xb[i] = (bf16_t)s;
}

// dt = softplus(dt_in @ dt_w^T + dt_b); K = dt_rank (3/6/12) -> VALU
__global__ void dt_softplus(const float* __restrict__ xdb, const float* __restrict__ dtw,
                            const float* __restrict__ dtb, float* __restrict__ dt,
                            int L, int di, int R) {
  int i = blockIdx.x * blockDim.x + threadIdx.x;
  if (i >= L * di) return;
  int l = i / di, c = i - l * di;
  float acc = dtb[c];
  for (int r = 0; r < R; ++r) acc += xdb[(size_t)l * 48 + r] * dtw[c * R + r];
  dt[i] = (acc > 20.f) ? acc : log1pf(__expf(acc));
}

// Selective-scan: lane = state n (0..15), 16-lane group = one channel.
__global__ __launch_bounds__(256)
void ssm_scan(const float* __restrict__ xf, const float* __restrict__ dt,
              const float* __restrict__ xdb, const float* __restrict__ A_log,
              float* __restrict__ y, int L, int di, int R) {
  const int n = threadIdx.x & 15;
  const int c = blockIdx.x * 16 + (threadIdx.x >> 4);
  if (c >= di) return;
  const float a = -__expf(A_log[c * 16 + n]);
  float h = 0.f;
  const float* Bp = xdb + R + n;
  const float* Cp = xdb + R + 16 + n;
  for (int l = 0; l < L; ++l) {
    const size_t o48 = (size_t)l * 48;
    const size_t od  = (size_t)l * di + c;
    // latency-bound loop: prefetch upcoming rows into cache
    __builtin_prefetch(xdb + o48 + 16 * 48, 0, 0);
    __builtin_prefetch(dt + od + 16 * di, 0, 0);
    const float dtv = dt[od];
    const float xv  = xf[od];
    const float bv  = Bp[o48];
    const float cv  = Cp[o48];
    h = __expf(dtv * a) * h + (dtv * bv) * xv;
    float yv = h * cv;
    yv += __shfl_xor(yv, 8, 32);
    yv += __shfl_xor(yv, 4, 32);
    yv += __shfl_xor(yv, 2, 32);
    yv += __shfl_xor(yv, 1, 32);
    if (n == 0) y[od] = yv;
  }
}

// y = (y_scan + x*D) * silu(z); write bf16 for out_proj GEMM
__global__ void gate_mul(const float* __restrict__ y, const float* __restrict__ xf,
                         const float* __restrict__ xz, const float* __restrict__ Dp,
                         bf16_t* __restrict__ ybf, int L, int di) {
  int i = blockIdx.x * blockDim.x + threadIdx.x;
  if (i >= L * di) return;
  int l = i / di, c = i - l * di;
  float zv = xz[(size_t)l * 2 * di + di + c];
  float sz = zv / (1.f + __expf(-zv));
  ybf[i] = (bf16_t)((y[i] + xf[i] * Dp[c]) * sz);
}

// enc output: seq (L,d) -> out (d,L), write-coalesced
__global__ void transpose_to_out(const float* __restrict__ seq, float* __restrict__ out,
                                 int L, int d) {
  int i = blockIdx.x * blockDim.x + threadIdx.x;
  if (i >= L * d) return;
  int c = i / L, l = i - c * L;
  out[i] = seq[(size_t)l * d + c];
}

// 3x3x3 stride-2 pad-1 conv, input (L,Ci) seq layout, output (L',Co) seq layout
__global__ void resconv(const float* __restrict__ seqIn, const float* __restrict__ w,
                        const float* __restrict__ b, float* __restrict__ seqOut,
                        int S, int O, int Ci, int Co) {
  int i = blockIdx.x * blockDim.x + threadIdx.x;
  if (i >= O * O * O * Co) return;
  int co = i % Co, ov = i / Co;
  int ow = ov % O, t = ov / O, oh = t % O, od = t / O;
  float acc = b[co];
  for (int ci = 0; ci < Ci; ++ci) {
    const float* wp = w + ((size_t)co * Ci + ci) * 27;
#pragma unroll
    for (int kd = 0; kd < 3; ++kd) {
      int id = 2 * od - 1 + kd; if ((unsigned)id >= (unsigned)S) continue;
#pragma unroll
      for (int kh = 0; kh < 3; ++kh) {
        int ih = 2 * oh - 1 + kh; if ((unsigned)ih >= (unsigned)S) continue;
#pragma unroll
        for (int kw = 0; kw < 3; ++kw) {
          int iw = 2 * ow - 1 + kw; if ((unsigned)iw >= (unsigned)S) continue;
          size_t iv = ((size_t)id * S + ih) * S + iw;
          acc += wp[kd * 9 + kh * 3 + kw] * seqIn[iv * Ci + ci];
        }
      }
    }
  }
  seqOut[i] = acc;
}

extern "C" void kernel_launch(void* const* d_in, const int* in_sizes, int n_in,
                              void* d_out, int out_size, void* d_ws, size_t ws_size,
                              hipStream_t stream) {
  (void)in_sizes; (void)n_in; (void)out_size; (void)ws_size;

  // --- input pointers (JAX pytree flatten order: dict keys sorted) ---
  const float* x_in = (const float*)d_in[0];
  const float* fus_b[3] = {(const float*)d_in[1], (const float*)d_in[3], (const float*)d_in[5]};
  const float* fus_w[3] = {(const float*)d_in[2], (const float*)d_in[4], (const float*)d_in[6]};
  const float* in_b = (const float*)d_in[7];
  const float* in_w = (const float*)d_in[8];
  const float* res_b[2] = {(const float*)d_in[9],  (const float*)d_in[11]};
  const float* res_w[2] = {(const float*)d_in[10], (const float*)d_in[12]};

  const int dims[3] = {48, 96, 192};
  const int Ls[3]   = {32768, 4096, 512};
  const int Ss[3]   = {32, 16, 8};

  // --- workspace carve (max sizes are stage 0) ---
  char* ws = (char*)d_ws;
  size_t off = 0;
  auto carve = [&](size_t bytes) -> void* {
    void* p = (void*)(ws + off);
    off += (bytes + 255) & ~(size_t)255;
    return p;
  };
  float*  seqA  = (float*) carve((size_t)32768 * 48 * 4);
  float*  seqB  = (float*) carve((size_t)4096 * 96 * 4);
  bf16_t* lnbf  = (bf16_t*)carve((size_t)32768 * 64 * 2);   // (L, Kp(d)) padded
  float*  xz    = (float*) carve((size_t)32768 * 192 * 4);
  float*  xf    = (float*) carve((size_t)32768 * 96 * 4);
  bf16_t* xbf   = (bf16_t*)carve((size_t)32768 * 96 * 2);
  float*  xdb   = (float*) carve((size_t)32768 * 48 * 4);
  float*  dtB   = (float*) carve((size_t)32768 * 96 * 4);
  float*  yB    = (float*) carve((size_t)32768 * 96 * 4);
  bf16_t* ybf   = (bf16_t*)carve((size_t)32768 * 96 * 2);
  bf16_t* wInB  = (bf16_t*)carve((size_t)768 * 192 * 2);
  bf16_t* wXB   = (bf16_t*)carve((size_t)48 * 384 * 2);
  bf16_t* wOutB = (bf16_t*)carve((size_t)192 * 384 * 2);
  bf16_t* wFusB = (bf16_t*)carve((size_t)192 * 192 * 2);

  float* outF = (float*)d_out;
  size_t encOff[3], skipOff[3], o = 0;
  for (int i = 0; i < 3; ++i) { encOff[i]  = o; o += (size_t)dims[i] * Ls[i]; }
  for (int i = 0; i < 3; ++i) { skipOff[i] = o; o += (size_t)dims[i] * Ls[i]; }

  // stem conv -> seq0 (L,48)
  { int n = Ls[0] * 48;
    init_seq<<<(n + 255) / 256, 256, 0, stream>>>(x_in, in_w, in_b, seqA, Ls[0]); }

  float* seqBufs[3] = {seqA, seqB, seqA};
  for (int st = 0; st < 3; ++st) {
    const int d = dims[st], L = Ls[st], di = 2 * d;
    const int R = (d + 15) / 16;           // dt_rank: 3 / 6 / 12
    const int KpD = (d + 31) & ~31;        // K padded to 32: 64 / 96 / 192
    float* seq = seqBufs[st];

    for (int blk = 0; blk < 2; ++blk) {
      const int pb = 13 + (st * 2 + blk) * 11;
      const float* A_log   = (const float*)d_in[pb + 0];
      const float* Dp      = (const float*)d_in[pb + 1];
      const float* convb   = (const float*)d_in[pb + 2];
      const float* convw   = (const float*)d_in[pb + 3];
      const float* dtb     = (const float*)d_in[pb + 4];
      const float* dtw     = (const float*)d_in[pb + 5];
      const float* inproj  = (const float*)d_in[pb + 6];
      const float* lnb     = (const float*)d_in[pb + 7];
      const float* lnw     = (const float*)d_in[pb + 8];
      const float* outproj = (const float*)d_in[pb + 9];
      const float* xproj   = (const float*)d_in[pb + 10];

      layernorm_bf16<<<L / 8, 256, 0, stream>>>(seq, lnw, lnb, lnbf, L, d, KpD);

      // xz = ln @ in_proj^T
      { int n = 2 * di * KpD;
        pack_bf16<<<(n + 255) / 256, 256, 0, stream>>>(inproj, wInB, 2 * di, d, 2 * di, KpD); }
      { dim3 g(L / 64, (2 * di) / 16);
        gemm_wmma_bf16<0><<<g, 128, 0, stream>>>(lnbf, wInB, xz, nullptr, KpD, 2 * di, 0); }

      { int n = L * di;
        dwconv_silu<<<(n + 255) / 256, 256, 0, stream>>>(xz, convw, convb, xf, xbf, L, di); }

      // xdb = x @ x_proj^T  (N=R+32 padded to 48 zero rows)
      { int n = 48 * di;
        pack_bf16<<<(n + 255) / 256, 256, 0, stream>>>(xproj, wXB, R + 32, di, 48, di); }
      { dim3 g(L / 64, 3);
        gemm_wmma_bf16<0><<<g, 128, 0, stream>>>(xbf, wXB, xdb, nullptr, di, 48, 0); }

      { int n = L * di;
        dt_softplus<<<(n + 255) / 256, 256, 0, stream>>>(xdb, dtw, dtb, dtB, L, di, R); }

      ssm_scan<<<di / 16, 256, 0, stream>>>(xf, dtB, xdb, A_log, yB, L, di, R);

      { int n = L * di;
        gate_mul<<<(n + 255) / 256, 256, 0, stream>>>(yB, xf, xz, Dp, ybf, L, di); }

      // seq += y @ out_proj^T
      { int n = d * di;
        pack_bf16<<<(n + 255) / 256, 256, 0, stream>>>(outproj, wOutB, d, di, d, di); }
      { dim3 g(L / 64, d / 16);
        gemm_wmma_bf16<1><<<g, 128, 0, stream>>>(ybf, wOutB, seq, nullptr, di, d, 0); }
    }

    // enc output: (L,d) -> (d,L)
    { int n = d * L;
      transpose_to_out<<<(n + 255) / 256, 256, 0, stream>>>(seq, outF + encOff[st], L, d); }

    // skip = fus 1x1 conv: WMMA, bias + transposed store straight into d_out
    { int n = d * KpD;
      pack_bf16<<<(n + 255) / 256, 256, 0, stream>>>(fus_w[st], wFusB, d, d, d, KpD); }
    { int n = L * KpD;
      pack_bf16<<<(n + 255) / 256, 256, 0, stream>>>(seq, lnbf, L, d, L, KpD); }
    { dim3 g(L / 64, d / 16);
      gemm_wmma_bf16<2><<<g, 128, 0, stream>>>(lnbf, wFusB, outF + skipOff[st],
                                               fus_b[st], KpD, 0, L); }

    // downsample to next stage
    if (st < 2) {
      const int Co = dims[st + 1], O = Ss[st + 1], S = Ss[st];
      int n = O * O * O * Co;
      resconv<<<(n + 255) / 256, 256, 0, stream>>>(seq, res_w[st], res_b[st],
                                                   seqBufs[st + 1], S, O, d, Co);
    }
  }
}
